// DenseDilatedKnnGraph_8031588843840
// MI455X (gfx1250) — matrix operand order repaired and checked by reading, not compiled, not run on hardware
//
#include <hip/hip_runtime.h>
#include <hip/hip_bf16.h>

typedef __attribute__((ext_vector_type(16))) __bf16 v16bf;
typedef __attribute__((ext_vector_type(8)))  __bf16 v8bf;
typedef __attribute__((ext_vector_type(8)))  float  v8f;

#define BDIM 4
#define CDIM 128
#define NDIM 8192
#define KKNN 36
#define KOUT 9
#define MROWS 32           // query rows per block (2 WMMA row tiles)
#define NB (NDIM / 128)    // key chunks of 128 columns

// ---------------------------------------------------------------------------
// Kernel 1: L2-normalize over channel dim, transpose (B,C,N) -> (B,N,C),
// convert to bf16, and record |y_n|^2 (post-normalization) per key column.
// ---------------------------------------------------------------------------
__global__ __launch_bounds__(256) void nrm_kernel(const float* __restrict__ x,
                                                  const float* __restrict__ y,
                                                  __bf16* __restrict__ Q,
                                                  __bf16* __restrict__ Kb,
                                                  float* __restrict__ ysq) {
  const int t = blockIdx.x * 256 + threadIdx.x;
  if (t >= BDIM * NDIM) return;
  const int b = t / NDIM;
  const int n = t - b * NDIM;
  const size_t xin = (size_t)b * CDIM * NDIM + n;  // x[((b*C + c)*N) + n]

  float sx = 0.f, sy = 0.f;
  for (int c = 0; c < CDIM; ++c) {
    const float vx = x[xin + (size_t)c * NDIM];
    const float vy = y[xin + (size_t)c * NDIM];
    sx += vx * vx;
    sy += vy * vy;
  }
  const float ix = 1.0f / fmaxf(sqrtf(sx), 1e-12f);
  const float iy = 1.0f / fmaxf(sqrtf(sy), 1e-12f);

  __bf16* qr = Q + (size_t)t * CDIM;
  __bf16* kr = Kb + (size_t)t * CDIM;
  for (int c = 0; c < CDIM; ++c) {
    qr[c] = (__bf16)(x[xin + (size_t)c * NDIM] * ix);
    kr[c] = (__bf16)(y[xin + (size_t)c * NDIM] * iy);
  }
  ysq[t] = sy * iy * iy;  // == 1 unless the clamp fired; kept for fidelity
}

// ---------------------------------------------------------------------------
// Kernel 2: fused bf16 WMMA GEMM (32 rows x 8192 keys, K=128) + streaming
// top-36 per row + dilated stride-4 index emission.
// Block: 256 threads = 8 wave32s. Wave w owns cols [w*16, w*16+16) of each
// 128-column chunk and computes TWO 16x16 row tiles sharing one set of B
// fragments (A lives in VGPRs for the whole sweep -> 32 FLOP per B byte).
// B fragments are double-buffered across chunks to overlap global loads
// with WMMA + the LDS top-k phase.
// ---------------------------------------------------------------------------
__global__ __launch_bounds__(256) void knn_kernel(const __bf16* __restrict__ Q,
                                                  const __bf16* __restrict__ Kb,
                                                  const float* __restrict__ ysq,
                                                  int* __restrict__ out) {
  __shared__ float tile[MROWS * 128];       // score tile: [row][col_local]
  __shared__ float tval[MROWS][KKNN + 2];   // sorted descending top-36 values
  __shared__ int   tidx[MROWS][KKNN + 2];   // matching key indices

  const int b    = blockIdx.y;
  const int m0   = blockIdx.x * MROWS;
  const int tid  = threadIdx.x;
  const int wv   = tid >> 5;
  const int lane = tid & 31;
  const int ln   = lane & 15;
  const int hi   = lane >> 4;

  for (int i = tid; i < MROWS * KKNN; i += 256) {
    tval[i / KKNN][i % KKNN] = -3.0e38f;
    tidx[i / KKNN][i % KKNN] = 0;
  }

  // ---- Load A fragments once (reused across the entire key sweep).
  // 16-bit A 16x32 layout: lane (hi=0): K = {0..7, 16..23}; (hi=1): +8.
  v16bf afr[2][4];
#pragma unroll
  for (int rt = 0; rt < 2; ++rt) {
    const __bf16* qrow = Q + ((size_t)b * NDIM + m0 + rt * 16 + ln) * CDIM;
#pragma unroll
    for (int ks = 0; ks < 4; ++ks) {
      const int base = ks * 32 + hi * 8;
      const v8bf c0 = *(const v8bf*)(qrow + base);
      const v8bf c1 = *(const v8bf*)(qrow + base + 16);
#pragma unroll
      for (int i = 0; i < 8; ++i) { afr[rt][ks][i] = c0[i]; afr[rt][ks][i + 8] = c1[i]; }
    }
  }

  const __bf16* kbase = Kb + (size_t)b * NDIM * CDIM;
  const float*  ybase = ysq + (size_t)b * NDIM;

  // 16-bit B 32x16 layout: lane = column, lanes 0-15 hold K={0..7, 16..23},
  // lanes 16-31 hold K={8..15, 24..31} (matches A).
  auto loadB = [&](int nb, v8bf* c) {
    const __bf16* kr = kbase + (size_t)(nb * 128 + wv * 16 + ln) * CDIM;
#pragma unroll
    for (int ks = 0; ks < 4; ++ks) {
      const int base = ks * 32 + hi * 8;
      c[2 * ks]     = *(const v8bf*)(kr + base);
      c[2 * ks + 1] = *(const v8bf*)(kr + base + 16);
    }
  };

  auto compute = [&](int nb, const v8bf* c) {
    v8f acc0 = {}, acc1 = {};
#pragma unroll
    for (int ks = 0; ks < 4; ++ks) {
      v16bf bfr;
#pragma unroll
      for (int i = 0; i < 8; ++i) { bfr[i] = c[2 * ks][i]; bfr[i + 8] = c[2 * ks + 1][i]; }
      acc0 = __builtin_amdgcn_wmma_f32_16x16x32_bf16(
          false, afr[0][ks], false, bfr, (short)0, acc0, false, false);
      acc1 = __builtin_amdgcn_wmma_f32_16x16x32_bf16(
          false, afr[1][ks], false, bfr, (short)0, acc1, false, false);
    }

    // score = 2*inner - |y|^2  (maximize <=> minimize distance; x_sq is
    // row-constant and cannot change the per-row top-k).
    const int cl = wv * 16 + ln;
    const float ys = ybase[nb * 128 + cl];
#pragma unroll
    for (int r = 0; r < 8; ++r) {
      tile[(hi * 8 + r) * 128 + cl]      = 2.0f * acc0[r] - ys;
      tile[(16 + hi * 8 + r) * 128 + cl] = 2.0f * acc1[r] - ys;
    }
    __syncthreads();

    // Streaming top-36 maintenance: one thread per row, threshold-gated
    // insertion sort (rarely taken after warm-up).
    if (tid < MROWS) {
      float th = tval[tid][KKNN - 1];
      const int g0 = nb * 128;
      for (int cc = 0; cc < 128; ++cc) {
        const float s = tile[tid * 128 + cc];
        if (s > th) {
          int p = KKNN - 1;
          while (p > 0 && tval[tid][p - 1] < s) {
            tval[tid][p] = tval[tid][p - 1];
            tidx[tid][p] = tidx[tid][p - 1];
            --p;
          }
          tval[tid][p] = s;
          tidx[tid][p] = g0 + cc;
          th = tval[tid][KKNN - 1];
        }
      }
    }
    __syncthreads();
  };

  // ---- Double-buffered sweep over all key chunks.
  v8bf bufA[8], bufB[8];
  loadB(0, bufA);
  for (int nb = 0; nb < NB; nb += 2) {
    loadB(nb + 1, bufB);              // in flight during compute(bufA)
    compute(nb, bufA);
    if (nb + 2 < NB) loadB(nb + 2, bufA);  // in flight during compute(bufB)
    compute(nb + 1, bufB);
  }

  // Emit edge_index[:, :, :, ::4]  ->  (2, B, N, 9)
  if (tid < MROWS) {
    const int n = m0 + tid;
    const size_t base0 = ((size_t)b * NDIM + n) * KOUT;
    const size_t half  = (size_t)BDIM * NDIM * KOUT;
#pragma unroll
    for (int j = 0; j < KOUT; ++j) {
      out[base0 + j]        = tidx[tid][j * 4];  // neighbor index
      out[half + base0 + j] = n;                 // center index
    }
  }
}

extern "C" void kernel_launch(void* const* d_in, const int* in_sizes, int n_in,
                              void* d_out, int out_size, void* d_ws, size_t ws_size,
                              hipStream_t stream) {
  const float* x = (const float*)d_in[0];
  const float* y = (const float*)d_in[1];

  char* ws = (char*)d_ws;
  const size_t qbytes = (size_t)BDIM * NDIM * CDIM * sizeof(__bf16);  // 8 MiB
  __bf16* Q   = (__bf16*)ws;
  __bf16* Kb  = (__bf16*)(ws + qbytes);
  float*  ysq = (float*)(ws + 2 * qbytes);  // + 128 KiB

  nrm_kernel<<<(BDIM * NDIM + 255) / 256, 256, 0, stream>>>(x, y, Q, Kb, ysq);

  dim3 grid(NDIM / MROWS, BDIM);
  knn_kernel<<<grid, 256, 0, stream>>>(Q, Kb, ysq, (int*)d_out);
}
